// StructurePredictionModule_54219667145445
// MI455X (gfx1250) — compile-verified, matrix operand order checked by reading, not run on hardware
//
#include <hip/hip_runtime.h>
#include <math.h>
#include <stdint.h>

// ---------------- problem constants ----------------
#define B_   2
#define L_   384
#define PD   64     // PAIR_DIM
#define HID  128
#define NDB  64     // N_DIST_BINS
#define NA   3      // N_ANGLES
#define NAB  36     // N_ANGLE_BINS
#define NAO  108    // NA*NAB
#define NAP  112    // padded to 7 tiles of 16

#define WAVES   4
#define THREADS (WAVES * 32)
#define TPW     8                        // tiles processed per wave (amortizes weight load)
#define ROWS_PER_BLOCK (16 * WAVES * TPW)  // 512

// f16 workspace layout (element offsets)
#define W_DW1T 0                 // [HID][PD]  = 8192
#define W_DW2T 8192              // [NDB][HID] = 8192
#define W_AW1T 16384             // [HID][PD]  = 8192
#define W_AW2T 24576             // [NAP][HID] = 14336
#define W_TOTAL 38912            // halves (77824 bytes)

// d_out layout (float offsets), tuple order:
// coordinates, distances, distance_logits, angles, angle_probs
#define OUT_COORD 0
#define OUT_DIST  (B_ * L_ * 3)                       // 2304
#define OUT_DLOG  (OUT_DIST + B_ * L_ * L_)           // 297216
#define OUT_ANG   (OUT_DLOG + B_ * L_ * L_ * NDB)     // 19171584
#define OUT_APROB (OUT_ANG + B_ * L_ * L_ * NA)       // 20056320

// per-wave LDS scratch layout (bytes)
#define SM_PAIR   0                      // 16*64  f16 = 2048
#define SM_H      2048                   // 16*128 f16 = 4096
#define SM_LOG    (2048 + 4096)          // 16*112 f32 = 7168 (dist uses 16*64)
#define SM_RED    (2048 + 4096 + 7168)   // 48*2   f32 = 384
#define SM_PERWAVE 13696
#define SM_TOTAL  (W_TOTAL * 2 + WAVES * SM_PERWAVE)  // 132608 bytes

typedef _Float16 v8h  __attribute__((ext_vector_type(8)));
typedef _Float16 v16h __attribute__((ext_vector_type(16)));
typedef float    v8f  __attribute__((ext_vector_type(8)));

// Build a 16x32 f16 A/B fragment for v_wmma_f32_16x16x32_f16.
// Per the CDNA5 ISA layout, a lane holds two contiguous 8-element K runs:
//   K = kbase + half*8 + [0,8)  and  K = kbase + 16 + half*8 + [0,8)
__device__ __forceinline__ v16h make_frag(const _Float16* __restrict__ rowBase,
                                          int kbase, int half) {
  v8h lo = *(const v8h*)(rowBase + kbase + half * 8);
  v8h hi = *(const v8h*)(rowBase + kbase + 16 + half * 8);
  return __builtin_shufflevector(lo, hi, 0, 1, 2, 3, 4, 5, 6, 7,
                                 8, 9, 10, 11, 12, 13, 14, 15);
}

__device__ __forceinline__ void lds_fence() {
  asm volatile("s_wait_dscnt 0" ::: "memory");
}

// ---------------- weight prep: f32 -> f16 transposed (N-major) ----------------
__global__ __launch_bounds__(256) void sp_prep(const float* __restrict__ dw1,
                                               const float* __restrict__ dw2,
                                               const float* __restrict__ aw1,
                                               const float* __restrict__ aw2,
                                               _Float16* __restrict__ wbuf) {
  int i = blockIdx.x * blockDim.x + threadIdx.x;
  if (i < W_DW2T) {                       // dw1T[n][k] = dw1[k][n], (64x128)->(128x64)
    int j = i - W_DW1T, n = j >> 6, k = j & 63;
    wbuf[i] = (_Float16)dw1[k * HID + n];
  } else if (i < W_AW1T) {                // dw2T[n][k] = dw2[k][n], (128x64)->(64x128)
    int j = i - W_DW2T, n = j >> 7, k = j & 127;
    wbuf[i] = (_Float16)dw2[k * NDB + n];
  } else if (i < W_AW2T) {                // aw1T[n][k] = aw1[k][n]
    int j = i - W_AW1T, n = j >> 6, k = j & 63;
    wbuf[i] = (_Float16)aw1[k * HID + n];
  } else if (i < W_TOTAL) {               // aw2T[n][k], pad n=108..111 with 0
    int j = i - W_AW2T, n = j >> 7, k = j & 127;
    wbuf[i] = (_Float16)((n < NAO) ? aw2[k * NAO + n] : 0.0f);
  }
}

// ---------------- fused main kernel ----------------
__global__ __launch_bounds__(THREADS) void sp_main(const float* __restrict__ pair,
                                                   const float* __restrict__ db1,
                                                   const float* __restrict__ db2,
                                                   const float* __restrict__ ab1,
                                                   const float* __restrict__ ab2,
                                                   const _Float16* __restrict__ wbuf,
                                                   float* __restrict__ out) {
  extern __shared__ char smem[];
  _Float16* sW = (_Float16*)smem;

  const int tid  = threadIdx.x;
  const int wave = tid >> 5;
  const int lane = tid & 31;
  const int half = lane >> 4;
  const int l16  = lane & 15;

  // cooperative weight load into LDS (77824 B = 4864 uint4)
  {
    const uint4* src = (const uint4*)wbuf;
    uint4* dst = (uint4*)sW;
    for (int i = tid; i < W_TOTAL / 8; i += THREADS) dst[i] = src[i];
  }
  __syncthreads();

  char* wb = smem + W_TOTAL * 2 + wave * SM_PERWAVE;
  _Float16* sPair = (_Float16*)(wb + SM_PAIR);
  _Float16* sH    = (_Float16*)(wb + SM_H);
  float*    sLog  = (float*)(wb + SM_LOG);
  float*    sRed  = (float*)(wb + SM_RED);

  const _Float16* dw1T = sW + W_DW1T;
  const _Float16* dw2T = sW + W_DW2T;
  const _Float16* aw1T = sW + W_AW1T;
  const _Float16* aw2T = sW + W_AW2T;

  const _Float16* aRow = sPair + l16 * PD;
  const _Float16* hRow = sH + l16 * HID;

  for (int t = 0; t < TPW; ++t) {
    // block's waves work on consecutive tiles each iteration (contiguous 64 rows)
    const int tile    = blockIdx.x * (WAVES * TPW) + t * WAVES + wave;
    const int rowbase = tile * 16;

    // prefetch next iteration's pair tile (global_prefetch_b8)
    if (t + 1 < TPW) {
      const float* np = pair + (size_t)(rowbase + 16 * WAVES) * PD + lane * 32;
      __builtin_prefetch(np, 0, 1);
    }

    // ---- load 16x64 pair tile, convert to f16 in LDS ----
    {
      const float4* src = (const float4*)(pair + (size_t)rowbase * PD);
      for (int u = 0; u < 8; ++u) {
        float4 v = src[lane + u * 32];
        int o = (lane + u * 32) * 4;
        sPair[o + 0] = (_Float16)v.x;
        sPair[o + 1] = (_Float16)v.y;
        sPair[o + 2] = (_Float16)v.z;
        sPair[o + 3] = (_Float16)v.w;
      }
    }
    lds_fence();

    // =========== distance head ===========
    // GEMM1: [16x64] x [64x128] -> relu -> sH (f16)
    for (int nt = 0; nt < 8; ++nt) {
      v8f acc = {};
      const _Float16* bRow = dw1T + (nt * 16 + l16) * PD;
      for (int kt = 0; kt < 2; ++kt) {
        v16h a = make_frag(aRow, kt * 32, half);
        v16h b = make_frag(bRow, kt * 32, half);
        acc = __builtin_amdgcn_wmma_f32_16x16x32_f16(false, a, false, b,
                                                     (short)0, acc, false, false);
      }
      float bias = db1[nt * 16 + l16];
      for (int r = 0; r < 8; ++r) {
        float v = acc[r] + bias;
        sH[(r + 8 * half) * HID + nt * 16 + l16] = (_Float16)fmaxf(v, 0.0f);
      }
    }
    lds_fence();

    // GEMM2: [16x128] x [128x64] -> logits (+bias) -> sLog (f32, stride 64)
    for (int nt = 0; nt < 4; ++nt) {
      v8f acc = {};
      const _Float16* bRow = dw2T + (nt * 16 + l16) * HID;
      for (int kt = 0; kt < 4; ++kt) {
        v16h a = make_frag(hRow, kt * 32, half);
        v16h b = make_frag(bRow, kt * 32, half);
        acc = __builtin_amdgcn_wmma_f32_16x16x32_f16(false, a, false, b,
                                                     (short)0, acc, false, false);
      }
      float bias = db2[nt * 16 + l16];
      for (int r = 0; r < 8; ++r)
        sLog[(r + 8 * half) * NDB + nt * 16 + l16] = acc[r] + bias;
    }
    lds_fence();

    // per-row softmax stats + expected distance (DIST_BINS = linspace(2,22,64))
    if (lane < 16) {
      const float* lr = sLog + lane * NDB;
      float m = -1e30f;
      for (int n = 0; n < NDB; ++n) m = fmaxf(m, lr[n]);
      float s = 0.0f, d = 0.0f;
      for (int n = 0; n < NDB; ++n) {
        float e = __expf(lr[n] - m);
        s += e;
        d += e * (2.0f + 20.0f * (float)n / 63.0f);
      }
      out[OUT_DIST + rowbase + lane] = d / s;
      sRed[lane * 2 + 0] = m;
      sRed[lane * 2 + 1] = 1.0f / s;
    }
    lds_fence();

    // coalesced softmax-prob writeout (sLog stride == NDB, so flat index works)
    for (int i = lane; i < 16 * NDB; i += 32) {
      int row = i >> 6;
      float p = __expf(sLog[i] - sRed[row * 2]) * sRed[row * 2 + 1];
      out[OUT_DLOG + rowbase * NDB + i] = p;
    }
    lds_fence();

    // =========== angle head ===========
    // GEMM3: [16x64] x [64x128] -> relu -> sH
    for (int nt = 0; nt < 8; ++nt) {
      v8f acc = {};
      const _Float16* bRow = aw1T + (nt * 16 + l16) * PD;
      for (int kt = 0; kt < 2; ++kt) {
        v16h a = make_frag(aRow, kt * 32, half);
        v16h b = make_frag(bRow, kt * 32, half);
        acc = __builtin_amdgcn_wmma_f32_16x16x32_f16(false, a, false, b,
                                                     (short)0, acc, false, false);
      }
      float bias = ab1[nt * 16 + l16];
      for (int r = 0; r < 8; ++r) {
        float v = acc[r] + bias;
        sH[(r + 8 * half) * HID + nt * 16 + l16] = (_Float16)fmaxf(v, 0.0f);
      }
    }
    lds_fence();

    // GEMM4: [16x128] x [128x112] -> logits (+bias) -> sLog (f32, stride 112)
    for (int nt = 0; nt < 7; ++nt) {
      v8f acc = {};
      const _Float16* bRow = aw2T + (nt * 16 + l16) * HID;
      for (int kt = 0; kt < 4; ++kt) {
        v16h a = make_frag(hRow, kt * 32, half);
        v16h b = make_frag(bRow, kt * 32, half);
        acc = __builtin_amdgcn_wmma_f32_16x16x32_f16(false, a, false, b,
                                                     (short)0, acc, false, false);
      }
      int n = nt * 16 + l16;
      float bias = (n < NAO) ? ab2[n] : 0.0f;
      for (int r = 0; r < 8; ++r)
        sLog[(r + 8 * half) * NAP + n] = acc[r] + bias;
    }
    lds_fence();

    // per (row, angle-chunk) softmax stats + expected angle
    // ANGLE_BINS = linspace(-180,180,36)
    for (int q = lane; q < 16 * NA; q += 32) {
      int row = q / NA, ch = q % NA;
      const float* lr = sLog + row * NAP + ch * NAB;
      float m = -1e30f;
      for (int n = 0; n < NAB; ++n) m = fmaxf(m, lr[n]);
      float s = 0.0f, a = 0.0f;
      for (int n = 0; n < NAB; ++n) {
        float e = __expf(lr[n] - m);
        s += e;
        a += e * (-180.0f + 360.0f * (float)n / 35.0f);
      }
      out[OUT_ANG + (rowbase + row) * NA + ch] = a / s;
      sRed[q * 2 + 0] = m;
      sRed[q * 2 + 1] = 1.0f / s;
    }
    lds_fence();

    // coalesced angle-prob writeout
    for (int i = lane; i < 16 * NAO; i += 32) {
      int row = i / NAO, n = i - row * NAO, ch = n / NAB;
      float p = __expf(sLog[row * NAP + n] - sRed[(row * NA + ch) * 2]) *
                sRed[(row * NA + ch) * 2 + 1];
      out[OUT_APROB + rowbase * NAO + i] = p;
    }
    lds_fence();
  }
}

// ---------------- coordinate construction ----------------
__global__ __launch_bounds__(L_) void sp_coords(const float* __restrict__ dist,
                                                float* __restrict__ coord) {
  int b = blockIdx.x;
  int i = threadIdx.x;
  if (i >= L_) return;
  const float* D = dist + (size_t)b * L_ * L_;
  float d01 = D[1];  // distances[b,0,1]
  float* c = coord + (b * L_ + i) * 3;
  if (i == 0) {
    c[0] = 0.0f; c[1] = 0.0f; c[2] = 0.0f;
  } else if (i == 1) {
    c[0] = d01; c[1] = 0.0f; c[2] = 0.0f;
  } else {
    float d0i = D[i];            // distances[b,0,i]
    float d1i = D[L_ + i];       // distances[b,1,i]
    float cs = (d01 * d01 + d0i * d0i - d1i * d1i) / (2.0f * d01 * d0i + 1e-8f);
    cs = fminf(1.0f, fmaxf(-1.0f, cs));
    float x = d0i * cs;
    float y = d0i * sqrtf(1.0f - cs * cs + 1e-8f);
    // deterministic pseudo-normal noise (stand-in for jax threefry key 42)
    uint32_t h = (uint32_t)(b * L_ + i) * 747796405u + 2891336453u;
    h ^= h >> 17; h *= 0xed5ad4bbu; h ^= h >> 11; h *= 0xac4c1b51u; h ^= h >> 15;
    uint32_t h2 = h * 0x31848babu; h2 ^= h2 >> 14;
    float u1 = (float)(h & 0xffffffu) * (1.0f / 16777216.0f) + 1e-7f;
    float u2 = (float)(h2 & 0xffffffu) * (1.0f / 16777216.0f);
    float z = 0.1f * sqrtf(-2.0f * __logf(u1)) * __cosf(6.2831853f * u2);
    c[0] = x; c[1] = y; c[2] = z;
  }
}

extern "C" void kernel_launch(void* const* d_in, const int* in_sizes, int n_in,
                              void* d_out, int out_size, void* d_ws, size_t ws_size,
                              hipStream_t stream) {
  const float* pair = (const float*)d_in[0];
  const float* dw1  = (const float*)d_in[1];
  const float* db1  = (const float*)d_in[2];
  const float* dw2  = (const float*)d_in[3];
  const float* db2  = (const float*)d_in[4];
  const float* aw1  = (const float*)d_in[5];
  const float* ab1  = (const float*)d_in[6];
  const float* aw2  = (const float*)d_in[7];
  const float* ab2  = (const float*)d_in[8];
  float* out = (float*)d_out;
  _Float16* wbuf = (_Float16*)d_ws;

  (void)in_sizes; (void)n_in; (void)out_size; (void)ws_size;

  static bool attr_set = false;  // idempotent, deterministic
  if (!attr_set) {
    hipFuncSetAttribute((const void*)sp_main,
                        hipFuncAttributeMaxDynamicSharedMemorySize, SM_TOTAL);
    attr_set = true;
  }

  // 1) convert weights to f16 transposed layouts in workspace
  sp_prep<<<(W_TOTAL + 255) / 256, 256, 0, stream>>>(dw1, dw2, aw1, aw2, wbuf);

  // 2) fused WMMA MLP heads + softmaxes (294912 rows / 512 per block)
  int nblocks = (B_ * L_ * L_) / ROWS_PER_BLOCK;  // 576
  sp_main<<<nblocks, THREADS, SM_TOTAL, stream>>>(pair, db1, db2, ab1, ab2, wbuf, out);

  // 3) coordinates from distances
  sp_coords<<<B_, L_, 0, stream>>>(out + OUT_DIST, out + OUT_COORD);
}
